// SRUCell_63909113364680
// MI455X (gfx1250) — matrix operand order repaired and checked by reading, not compiled, not run on hardware
//
#include <hip/hip_runtime.h>

// ---------------------------------------------------------------------------
// SRUCell forward, MI455X (gfx1250, wave32, WMMA).
//
//   hy[b,h] = hx[b,h] + sum_j weight[h,j,id[b]]*hx[b,j] + bias[h,id[b]]
//
// Roofline: weight = 134 MB read-once @ 23.3 TB/s => ~5.8 us floor; all other
// operands are KB-scale.  Dense GEMM  rec^T[h,b] = sum_k W[h,k] * X[k,b] with
// K = 32768, X[j*32+i, b] = hx[b,j]*(i==id[b]) built per 32-wide K-step in
// LDS (one j per step).  bf16 v_wmma_f32_16x16x32_bf16; 32x FLOP redundancy
// (~8.6 GF) stays under the memory floor at CDNA5 bf16 matrix rates.
// Split-K x64 (512 WGs x 8 waves) for WGP occupancy; f32 atomics fold the
// partials into an output pre-seeded with hx + bias.
//
// K-permutation trick: A is loaded as 16 *contiguous* floats per lane; the
// resulting (ISA slot -> logical k) map is "swap bit3<->bit4", and B is read
// the same contiguous way from LDS, so both operands pair identical logical k.
// ---------------------------------------------------------------------------

typedef __attribute__((ext_vector_type(16))) __bf16 v16bf;
typedef __attribute__((ext_vector_type(16))) float  v16f;
typedef __attribute__((ext_vector_type(8)))  float  v8f;

#define HID    1024
#define BATCH  128
#define ISZ    32
#define KTOT   (HID * ISZ)        // 32768
#define KCHUNK 512                // K elements per workgroup
#define NSPLIT (KTOT / KCHUNK)    // 64 split-K chunks
#define STEPS  (KCHUNK / 32)      // 16 WMMA K-steps per chunk
#define STAGE  4                  // K-steps staged in LDS per build phase

__device__ inline unsigned short bf16bits(float x) {
    union { __bf16 b; unsigned short u; } r;
    r.b = (__bf16)x;
    return r.u;
}

// out[b*H+h] = hx[b*H+h] + bias[h*32 + argmax_i input[b,i]]
__global__ __launch_bounds__(256) void sru_init(
    const float* __restrict__ inp, const float* __restrict__ hx,
    const float* __restrict__ bias, float* __restrict__ out) {
    int t = blockIdx.x * 256 + threadIdx.x;     // 0 .. 131071
    int b = t >> 10;
    int h = t & (HID - 1);
    const float* row = inp + b * ISZ;
    float best = row[0]; int id = 0;
    #pragma unroll
    for (int i = 1; i < ISZ; ++i) { float v = row[i]; if (v > best) { best = v; id = i; } }
    out[t] = hx[t] + bias[h * ISZ + id];
}

__global__ __launch_bounds__(256) void sru_gemm(
    const float* __restrict__ inp, const float* __restrict__ hx,
    const float* __restrict__ W,   float* __restrict__ out) {
    // B-tile staging: [stage s][batch b][k-slot] bf16, k contiguous (32 KB)
    __shared__ unsigned short Bt[STAGE][BATCH][ISZ];
    __shared__ int sidx[BATCH];

    const int tid  = threadIdx.x;
    const int wv   = tid >> 5;          // wave 0..7
    const int lane = tid & 31;
    const int h0   = blockIdx.y * 128 + wv * 16;   // this wave's 16 h-rows
    const int j0   = blockIdx.x * STEPS;           // 16 j-values per k-chunk

    if (tid < BATCH) {                  // one-hot -> index
        const float* row = inp + tid * ISZ;
        float best = row[0]; int id = 0;
        #pragma unroll
        for (int i = 1; i < ISZ; ++i) { float v = row[i]; if (v > best) { best = v; id = i; } }
        sidx[tid] = id;
    }

    const int m    = lane & 15;                // A row / B col within tile
    const int kb16 = (lane < 16) ? 0 : 16;     // contiguous 16-float half per lane
    const float* wrow = W + (size_t)(h0 + m) * KTOT + (size_t)j0 * 32 + kb16;

    v8f acc[8];
    #pragma unroll
    for (int g = 0; g < 8; ++g) acc[g] = (v8f){0.f,0.f,0.f,0.f,0.f,0.f,0.f,0.f};

    for (int outer = 0; outer < STEPS / STAGE; ++outer) {
        __syncthreads();                       // prev compute done, sidx visible
        // zero 32 KB stage buffer: 2048 uint4 over 256 threads
        {
            uint4 z = {0u, 0u, 0u, 0u};
            uint4* p = (uint4*)&Bt[0][0][0];
            #pragma unroll
            for (int q = 0; q < 8; ++q) p[tid + q * 256] = z;
        }
        __syncthreads();
        // scatter: Bt[s][b][id[b]] = bf16(hx[b, j0 + outer*STAGE + s])
        #pragma unroll
        for (int q = 0; q < 2; ++q) {
            int e = tid + q * 256;             // 0..511 = STAGE*BATCH
            int s = e >> 7;
            int b = e & (BATCH - 1);
            int j = j0 + outer * STAGE + s;
            int id = sidx[b];
            int pos = id;
            Bt[s][b][pos] = bf16bits(hx[b * HID + j]);
        }
        __syncthreads();

        #pragma unroll
        for (int s = 0; s < STAGE; ++s) {
            // ---- A fragment: 16 contiguous floats of W[h0+m, step*32 + kb16 ..]
            // loaded as 4x b128, converted in one shot (8x v_cvt_pk_bf16_f32).
            const float* ap = wrow + (size_t)(outer * STAGE + s) * 32;
            union { float4 q[4]; v16f v; } Af;
            Af.q[0] = *(const float4*)(ap + 0);
            Af.q[1] = *(const float4*)(ap + 4);
            Af.q[2] = *(const float4*)(ap + 8);
            Af.q[3] = *(const float4*)(ap + 12);
            v16bf A = __builtin_convertvector(Af.v, v16bf);

            // ---- B fragments: software-pipelined one group ahead so the
            // ds_load for g+1 is outstanding while WMMA g issues.
            const unsigned short* bp = &Bt[s][m][kb16];    // group stride 16*32
            uint4 f0 = *(const uint4*)(bp);
            uint4 f1 = *(const uint4*)(bp + 8);
            #pragma unroll
            for (int g = 0; g < 8; ++g) {
                union { uint4 q[2]; v16bf v; } Bf;
                Bf.q[0] = f0;
                Bf.q[1] = f1;
                if (g < 7) {
                    f0 = *(const uint4*)(bp + (g + 1) * 512);
                    f1 = *(const uint4*)(bp + (g + 1) * 512 + 8);
                }
                acc[g] = __builtin_amdgcn_wmma_f32_16x16x32_bf16(
                    false, A, false, Bf.v, (short)0, acc[g], false, false);
            }
        }
    }

    // Accumulate split-K partials: D[m,n] -> out[b*HID + h]
    const int hh = h0 + ((lane < 16) ? 0 : 8);
    #pragma unroll
    for (int g = 0; g < 8; ++g) {
        int b = g * 16 + m;
        #pragma unroll
        for (int r = 0; r < 8; ++r) {
            unsafeAtomicAdd(&out[b * HID + hh + r], acc[g][r]);
        }
    }
}

extern "C" void kernel_launch(void* const* d_in, const int* in_sizes, int n_in,
                              void* d_out, int out_size, void* d_ws, size_t ws_size,
                              hipStream_t stream) {
    (void)in_sizes; (void)n_in; (void)d_ws; (void)ws_size; (void)out_size;
    const float* inp  = (const float*)d_in[0];   // [128,32] one-hot
    const float* hx   = (const float*)d_in[1];   // [1,128,1024]
    const float* wgt  = (const float*)d_in[2];   // [1024,1024,32]
    const float* bias = (const float*)d_in[3];   // [1024,32]
    float* out = (float*)d_out;                  // [1,128,1024]

    // Seed out = hx + bias[h, id[b]]  (fully overwrites poisoned d_out)
    sru_init<<<(BATCH * HID) / 256, 256, 0, stream>>>(inp, hx, bias, out);
    // Split-K WMMA GEMM accumulates the recurrent term on top
    sru_gemm<<<dim3(NSPLIT, (HID / 128)), 256, 0, stream>>>(inp, hx, wgt, out);
}
